// KGEModel_54511724921273
// MI455X (gfx1250) — compile-verified, matrix operand order checked by reading, not run on hardware
//
#include <hip/hip_runtime.h>
#include <hip/hip_fp16.h>

typedef __attribute__((ext_vector_type(16))) _Float16 v16h;
typedef __attribute__((ext_vector_type(8)))  float    v8f;

#define EMB   256
#define H1    1024
#define H2    512
#define NENT  700
#define NDRUG 200
#define OUT_PAIRS 340000
#define DRUG_OUT_OFF (OUT_PAIRS * 2)
#define NTILES 10628   // 32-pair tiles over 10 pair-tables

// Pair tables: (dp, dd, ds, de, ps, pp, pd, ed, sp, sd)
__constant__ int c_Na[10]   = {200,200,200,200,200,200,200,150,150,150};
__constant__ int c_Nb[10]   = {200,200,150,150,150,200,200,200,200,200};
__constant__ int c_Aoff[10] = {0,0,0,0,200,200,200,550,400,400};
__constant__ int c_Boff[10] = {200,0,400,550,400,200,0,0,200,0};
__constant__ int c_Ooff[10] = {0,40000,80000,110000,140000,170000,210000,250000,280000,310000};
__constant__ int c_Toff[11] = {0,1250,2500,3438,4376,5314,6564,7814,8752,9690,10628};

// ------------------------------------------------------------------
// ent = entity_embedding @ W_ent  (700x256 * 256x256); also emit drug slice.
__global__ __launch_bounds__(256) void k_ent(const float* __restrict__ E,
                                             const float* __restrict__ Went,
                                             float* __restrict__ ent,
                                             float* __restrict__ out) {
  const int r = blockIdx.x;
  const int c = threadIdx.x;
  const float* er = E + r * EMB;
  float acc = 0.f;
  for (int k = 0; k < EMB; ++k) acc = fmaf(er[k], Went[k * EMB + c], acc);
  ent[r * EMB + c] = acc;
  if (r < NDRUG) out[DRUG_OUT_OFF + r * EMB + c] = acc;
}

// ------------------------------------------------------------------
// HA = ent @ W1[:256,:], HB = ent @ W1[256:,:]   (700x1024 each)
__global__ __launch_bounds__(256) void k_hab(const float* __restrict__ ent,
                                             const float* __restrict__ W1,
                                             float* __restrict__ HA,
                                             float* __restrict__ HB) {
  const int r = blockIdx.x;
  const int c = blockIdx.y * 256 + threadIdx.x;
  const float* er = ent + r * EMB;
  float a = 0.f, b = 0.f;
  for (int k = 0; k < EMB; ++k) {
    const float e = er[k];
    a = fmaf(e, W1[k * H1 + c], a);
    b = fmaf(e, W1[(EMB + k) * H1 + c], b);
  }
  HA[r * H1 + c] = a;
  HB[r * H1 + c] = b;
}

// ------------------------------------------------------------------
// Repack W2 (1024x512 f32, row-major) into f16 WMMA B-fragment order.
// Linear index bits: [nt(5)][kk(5)][lane(5)][half(4)]
// B layout (32x16 f16): n = lane%16 ; k = 32*kk + 16*(h/8) + 8*(lane/16) + (h%8)
__global__ __launch_bounds__(256) void k_pack(const float* __restrict__ W2,
                                              _Float16* __restrict__ W2p) {
  const int gid  = blockIdx.x * 256 + threadIdx.x;   // 0 .. 524287
  const int hh   = gid & 15;
  const int lane = (gid >> 4) & 31;
  const int kk   = (gid >> 9) & 31;
  const int nt   = gid >> 14;
  const int k = kk * 32 + ((hh >> 3) << 4) + ((lane >> 4) << 3) + (hh & 7);
  const int n = nt * 16 + (lane & 15);
  W2p[gid] = (_Float16)W2[k * H2 + n];
}

// ------------------------------------------------------------------
// Fused pair MLP: h=relu(ha+hb+b1) -> (WMMA f16) h@W2 +b2, relu ->
// h2@W3 + b3 -> softmax.  One block = 32 pairs x 512 columns.
// Each wave: 2 M-tiles x 4 N-tiles = 8 accumulators -> 8 WMMAs per 4 B loads.
__global__ __launch_bounds__(256) void k_pair(const float* __restrict__ HA,
                                              const float* __restrict__ HB,
                                              const float* __restrict__ b1,
                                              const _Float16* __restrict__ W2p,
                                              const float* __restrict__ b2,
                                              const float* __restrict__ W3,
                                              const float* __restrict__ b3,
                                              float* __restrict__ out) {
  __shared__ _Float16 sh_h[32 * H1];   // A fragments (2 M-tiles), 64 KB
  __shared__ float    sh_h2[32 * H2];  // hidden2, 64 KB
  __shared__ int      sh_ra[32];       // a-side row (or -1 if padding)
  __shared__ int      sh_rb[32];       // b-side row

  const int t = blockIdx.x;
  int p = 0;
  #pragma unroll
  for (int i = 1; i < 10; ++i) p += (t >= c_Toff[i]) ? 1 : 0;
  const int pbase = (t - c_Toff[p]) * 32;
  const int nb    = c_Nb[p];
  const int npair = c_Na[p] * nb;
  const int tid   = threadIdx.x;

  // Per-pair row precompute (hoists the integer divide out of stage A)
  if (tid < 32) {
    const int pm = pbase + tid;
    const int i  = pm / nb;
    const int j  = pm - i * nb;
    sh_ra[tid] = (pm < npair) ? (c_Aoff[p] + i) : -1;
    sh_rb[tid] = c_Boff[p] + j;
  }
  __syncthreads();

  // ---- Stage A: build h = relu(ha_i + hb_j + b1) in A-fragment LDS order
  for (int idx = tid; idx < 32 * H1; idx += 256) {
    const int m  = idx >> 10;        // pair slot 0..31
    const int k  = idx & (H1 - 1);
    const int ra = sh_ra[m];
    float v = 0.f;
    if (ra >= 0) {
      v = HA[ra * H1 + k] + HB[sh_rb[m] * H1 + k] + b1[k];
      v = v > 0.f ? v : 0.f;
    }
    // A-matrix 16x32 f16 VGPR layout (ISA 7.12.2); two M-tiles stacked
    const int mt   = m >> 4;
    const int ml   = m & 15;
    const int kk   = k >> 5;
    const int kl   = k & 31;
    const int lane = ml + (((kl & 15) >= 8) ? 16 : 0);
    const int hh   = ((kl >> 4) << 3) + (kl & 7);
    sh_h[mt * (16 * H1) + kk * 512 + lane * 16 + hh] = (_Float16)v;
  }
  __syncthreads();

  // ---- Stage B: (32x1024) x (1024x512) via v_wmma_f32_16x16x32_f16
  const int wave = tid >> 5;
  const int lane = tid & 31;
  v8f a00 = {}, a01 = {}, a02 = {}, a03 = {};   // M-tile 0, N-tiles wave+{0,8,16,24}
  v8f a10 = {}, a11 = {}, a12 = {}, a13 = {};   // M-tile 1
  const _Float16* ash0 = sh_h + lane * 16;
  const _Float16* ash1 = sh_h + 16 * H1 + lane * 16;
  const _Float16* bsh  = W2p + lane * 16;
  for (int kk = 0; kk < 32; ++kk) {
    const v16h fa0 = *(const v16h*)(ash0 + kk * 512);
    const v16h fa1 = *(const v16h*)(ash1 + kk * 512);
    const v16h fb0 = *(const v16h*)(bsh + ((wave      ) * 32 + kk) * 512);
    const v16h fb1 = *(const v16h*)(bsh + ((wave +  8) * 32 + kk) * 512);
    const v16h fb2 = *(const v16h*)(bsh + ((wave + 16) * 32 + kk) * 512);
    const v16h fb3 = *(const v16h*)(bsh + ((wave + 24) * 32 + kk) * 512);
    a00 = __builtin_amdgcn_wmma_f32_16x16x32_f16(false, fa0, false, fb0, (short)0, a00, false, false);
    a10 = __builtin_amdgcn_wmma_f32_16x16x32_f16(false, fa1, false, fb0, (short)0, a10, false, false);
    a01 = __builtin_amdgcn_wmma_f32_16x16x32_f16(false, fa0, false, fb1, (short)0, a01, false, false);
    a11 = __builtin_amdgcn_wmma_f32_16x16x32_f16(false, fa1, false, fb1, (short)0, a11, false, false);
    a02 = __builtin_amdgcn_wmma_f32_16x16x32_f16(false, fa0, false, fb2, (short)0, a02, false, false);
    a12 = __builtin_amdgcn_wmma_f32_16x16x32_f16(false, fa1, false, fb2, (short)0, a12, false, false);
    a03 = __builtin_amdgcn_wmma_f32_16x16x32_f16(false, fa0, false, fb3, (short)0, a03, false, false);
    a13 = __builtin_amdgcn_wmma_f32_16x16x32_f16(false, fa1, false, fb3, (short)0, a13, false, false);
  }
  // C/D layout: M = r + 8*(lane/16), N = lane%16
  const int mrow = (lane >> 4) * 8;
  const int ncol = lane & 15;
  #pragma unroll
  for (int r = 0; r < 8; ++r) {
    int n; float v;
    n = (wave      ) * 16 + ncol;
    v = a00[r] + b2[n]; sh_h2[(mrow + r) * H2 + n]        = v > 0.f ? v : 0.f;
    v = a10[r] + b2[n]; sh_h2[(16 + mrow + r) * H2 + n]   = v > 0.f ? v : 0.f;
    n = (wave +  8) * 16 + ncol;
    v = a01[r] + b2[n]; sh_h2[(mrow + r) * H2 + n]        = v > 0.f ? v : 0.f;
    v = a11[r] + b2[n]; sh_h2[(16 + mrow + r) * H2 + n]   = v > 0.f ? v : 0.f;
    n = (wave + 16) * 16 + ncol;
    v = a02[r] + b2[n]; sh_h2[(mrow + r) * H2 + n]        = v > 0.f ? v : 0.f;
    v = a12[r] + b2[n]; sh_h2[(16 + mrow + r) * H2 + n]   = v > 0.f ? v : 0.f;
    n = (wave + 24) * 16 + ncol;
    v = a03[r] + b2[n]; sh_h2[(mrow + r) * H2 + n]        = v > 0.f ? v : 0.f;
    v = a13[r] + b2[n]; sh_h2[(16 + mrow + r) * H2 + n]   = v > 0.f ? v : 0.f;
  }
  __syncthreads();

  // ---- Stage C: logits = h2 @ W3 + b3, softmax over 2 classes
  if (tid < 32) {
    const int m  = tid;
    const int pm = pbase + m;
    if (pm < npair) {
      float c0 = b3[0], c1 = b3[1];
      const float* hr = sh_h2 + m * H2;
      for (int k = 0; k < H2; ++k) {
        const float hv = hr[k];
        c0 = fmaf(hv, W3[2 * k + 0], c0);
        c1 = fmaf(hv, W3[2 * k + 1], c1);
      }
      const float mx = c0 > c1 ? c0 : c1;
      const float e0 = __expf(c0 - mx);
      const float e1 = __expf(c1 - mx);
      const float inv = 1.f / (e0 + e1);
      out[(c_Ooff[p] + pm) * 2 + 0] = e0 * inv;
      out[(c_Ooff[p] + pm) * 2 + 1] = e1 * inv;
    }
  }
}

// ------------------------------------------------------------------
extern "C" void kernel_launch(void* const* d_in, const int* in_sizes, int n_in,
                              void* d_out, int out_size, void* d_ws, size_t ws_size,
                              hipStream_t stream) {
  (void)in_sizes; (void)n_in; (void)out_size; (void)ws_size;
  // setup_inputs order: 0 type_mask, 1 entity_embedding, 2 relation_embedding,
  // 3 W_ent, 4 W_rel, 5 W1, 6 b1, 7 W2, 8 b2, 9 W3, 10 b3
  const float* E    = (const float*)d_in[1];
  const float* Went = (const float*)d_in[3];
  const float* W1   = (const float*)d_in[5];
  const float* b1   = (const float*)d_in[6];
  const float* W2   = (const float*)d_in[7];
  const float* b2   = (const float*)d_in[8];
  const float* W3   = (const float*)d_in[9];
  const float* b3   = (const float*)d_in[10];
  float* out = (float*)d_out;

  // Workspace layout (all 256B-aligned)
  float*    ent = (float*)d_ws;                 // 700*256 f32
  float*    HA  = ent + NENT * EMB;             // 700*1024 f32
  float*    HB  = HA + NENT * H1;               // 700*1024 f32
  _Float16* W2p = (_Float16*)(HB + NENT * H1);  // 1024*512 f16 (fragment order)

  k_ent<<<NENT, 256, 0, stream>>>(E, Went, ent, out);
  k_hab<<<dim3(NENT, H1 / 256), 256, 0, stream>>>(ent, W1, HA, HB);
  k_pack<<<(H1 * H2) / 256, 256, 0, stream>>>(W2, W2p);
  k_pair<<<NTILES, 256, 0, stream>>>(HA, HB, b1, W2p, b2, W3, b3, out);
}
